// FirstDotProdAttAggr_4045859192940
// MI455X (gfx1250) — matrix operand order repaired and checked by reading, not compiled, not run on hardware
//
#include <hip/hip_runtime.h>
#include <hip/hip_bf16.h>

typedef __attribute__((ext_vector_type(16))) _Float16 v16h;
typedef __attribute__((ext_vector_type(8)))  float    v8f;
typedef __attribute__((ext_vector_type(4)))  float    v4f;
typedef __attribute__((ext_vector_type(4)))  _Float16 v4h;

union V16H { v16h v; _Float16 e[16]; v4f f4[2]; };
union V8F  { v8f  v; float f[8]; };

#define N_ATOMS 20000
#define N_PAIRS 320000
#define F_DIM   128
#define N_HEAD  8
#define FH      16
#define M_TOT   8
#define K_RBF   32
#define F_HID   128
#define F_OUT   256   // N_DEG * F_DIM

// ---------------------------------------------------------------- utilities
__global__ void k_zero(float* __restrict__ p, int n) {
    for (int i = blockIdx.x * blockDim.x + threadIdx.x; i < n;
         i += gridDim.x * blockDim.x)
        p[i] = 0.f;
}

// Transpose + fp16-convert MLP weights into workspace: W0T[n][k], W1T[n][k]
__global__ void k_wt(const float* __restrict__ W0, const float* __restrict__ W1,
                     _Float16* __restrict__ W0T, _Float16* __restrict__ W1T) {
    const int total = K_RBF * F_HID + F_HID * F_OUT;
    for (int i = blockIdx.x * blockDim.x + threadIdx.x; i < total;
         i += gridDim.x * blockDim.x) {
        if (i < K_RBF * F_HID) {
            int n = i / K_RBF, k = i % K_RBF;
            W0T[i] = (_Float16)W0[k * F_HID + n];
        } else {
            int j = i - K_RBF * F_HID;
            int n = j / F_HID, k = j % F_HID;
            W1T[j] = (_Float16)W1[k * F_OUT + n];
        }
    }
}

// ------------------------------------------------- per-atom Q/K projections
// Q[n,d,h,a] = sum_b W_Q[d,h,a,b] * x[n, h*16+b]   (and same for K)
// One wave = 16 atoms. WMMA: A = x-tile (16x16 padded to K=32), B = W^T.
__global__ __launch_bounds__(128)
void k_qk(const float* __restrict__ x,
          const float* __restrict__ Wq, const float* __restrict__ Wk,
          float* __restrict__ Qws, float* __restrict__ Kws) {
    const int lane = threadIdx.x & 31;
    const int wave = threadIdx.x >> 5;
    const int tile = blockIdx.x * 4 + wave;        // 16 atoms per tile
    if (tile * 16 >= N_ATOMS) return;              // wave-uniform
    const int atom0 = tile * 16;
    const int hl  = lane >> 4;                     // half of wave
    const int l15 = lane & 15;

    for (int h = 0; h < N_HEAD; ++h) {
        // A fragment: rows = atoms, K = b (0..15 real, 16..31 zero pad)
        V16H A;
        {
            const float* xr = x + (atom0 + l15) * F_DIM + h * FH;
            #pragma unroll
            for (int k = 0; k < 16; ++k) A.e[k] = hl ? (_Float16)0.f : (_Float16)xr[k];
        }
        for (int d = 0; d < 2; ++d) {
            #pragma unroll
            for (int mat = 0; mat < 2; ++mat) {
                const float* W =
                    (mat == 0 ? Wq : Wk) + ((d * N_HEAD + h) * FH) * FH; // [a][b]
                // B[k=b][n=a] = W[a][b]; lanes>=16 carry K=16..31 -> zero
                V16H B;
                if (hl == 0) {
                    const float* wr = W + l15 * FH;
                    #pragma unroll
                    for (int k = 0; k < 16; ++k) B.e[k] = (_Float16)wr[k];
                } else {
                    #pragma unroll
                    for (int k = 0; k < 16; ++k) B.e[k] = (_Float16)0.f;
                }
                V8F C;
                #pragma unroll
                for (int r = 0; r < 8; ++r) C.f[r] = 0.f;
                C.v = __builtin_amdgcn_wmma_f32_16x16x32_f16(
                    false, A.v, false, B.v, (short)0, C.v, false, false);
                float* dst = (mat == 0 ? Qws : Kws);
                #pragma unroll
                for (int r = 0; r < 8; ++r) {
                    int m = hl ? (r + 8) : r;
                    dst[(atom0 + m) * 256 + d * 128 + h * FH + l15] = C.f[r];
                }
            }
        }
    }
}

// ----------------------------------------------------------- main pair loop
// One wave = 32 pairs (two 16-row WMMA tiles sharing every B fragment).
struct WaveScratch {
    _Float16 hbuf[32 * F_HID];   // layer-0 activations  [m][k]    8 KB
    _Float16 wij[32 * F_OUT];    // filter tile          [m][n]   16 KB
    float    alpha[32 * 16];     // [p][d*8+h]                     2 KB
    float    phi[32];
    int      ii[32], jj[32];
};                               // 26.375 KB per wave

__device__ __forceinline__ void flush_acc(float* __restrict__ out, int i,
                                          int lane, float acc[8][4]) {
    #pragma unroll
    for (int mm = 0; mm < 8; ++mm)
        #pragma unroll
        for (int s = 0; s < 4; ++s)
            atomicAdd(out + i * 1024 + mm * 128 + lane + 32 * s, acc[mm][s]);
}

__global__ __launch_bounds__(64)
void k_main(const float* __restrict__ rbf, const float* __restrict__ sph,
            const float* __restrict__ phi_r_cut,
            const int* __restrict__ idx_i, const int* __restrict__ idx_j,
            const float* __restrict__ pmask,
            const float* __restrict__ b0, const float* __restrict__ b1,
            const float* __restrict__ x,
            const float* __restrict__ Qws, const float* __restrict__ Kws,
            const _Float16* __restrict__ W0T, const _Float16* __restrict__ W1T,
            float* __restrict__ out) {
    __shared__ WaveScratch smem[2];
    const int lane = threadIdx.x & 31;
    const int wave = threadIdx.x >> 5;
    WaveScratch& S = smem[wave];
    const int hl  = lane >> 4;
    const int l15 = lane & 15;
    const int nSuper  = N_PAIRS / 32;              // 32 pairs per super-tile
    const int wstride = gridDim.x * 2;

    for (int st = blockIdx.x * 2 + wave; st < nSuper; st += wstride) {
        const int p0 = st * 32;
        S.phi[lane] = phi_r_cut[p0 + lane];
        S.ii[lane]  = idx_i[p0 + lane];
        S.jj[lane]  = idx_j[p0 + lane];

        // ---- layer 0: h = ssp(rbf @ W0 + b0), two 16-row tiles ------------
        V16H A0[2];   // rbf tiles, K = 32 exactly
        #pragma unroll
        for (int u = 0; u < 2; ++u) {
            const float* rr = rbf + (p0 + u * 16 + l15) * K_RBF + hl * 8;
            #pragma unroll
            for (int k = 0; k < 8; ++k) A0[u].e[k]     = (_Float16)rr[k];
            #pragma unroll
            for (int k = 0; k < 8; ++k) A0[u].e[8 + k] = (_Float16)rr[16 + k];
        }
        #pragma unroll 1
        for (int j = 0; j < 8; ++j) {
            V16H B;   // W0T[n][k]: per lane two 8-half runs, k = 8hl+0..7 and 16+8hl+0..7
            {
                const _Float16* wn = W0T + (j * 16 + l15) * K_RBF + hl * 8;
                B.f4[0] = *(const v4f*)wn;
                B.f4[1] = *(const v4f*)(wn + 16);
            }
            const float bb = b0[j * 16 + l15];
            #pragma unroll
            for (int u = 0; u < 2; ++u) {
                V8F C;
                #pragma unroll
                for (int r = 0; r < 8; ++r) C.f[r] = bb;
                C.v = __builtin_amdgcn_wmma_f32_16x16x32_f16(
                    false, A0[u].v, false, B.v, (short)0, C.v, false, false);
                #pragma unroll
                for (int r = 0; r < 8; ++r) {
                    int m = u * 16 + (hl ? r + 8 : r);
                    float y = C.f[r];
                    float s = fmaxf(y, 0.f) + log1pf(__expf(-fabsf(y)))
                              - 0.6931471805599453f;          // shifted softplus
                    S.hbuf[m * F_HID + j * 16 + l15] = (_Float16)s;
                }
            }
        }

        // ---- layer 1: W_ij = (h @ W1 + b1) * phi --------------------------
        V16H A1[2][4];   // K = 128 in 4 chunks, re-read through LDS
        #pragma unroll
        for (int u = 0; u < 2; ++u)
            #pragma unroll
            for (int c = 0; c < 4; ++c) {
                const _Float16* hr =
                    S.hbuf + (u * 16 + l15) * F_HID + c * 32 + hl * 8;
                A1[u][c].f4[0] = *(const v4f*)hr;
                A1[u][c].f4[1] = *(const v4f*)(hr + 16);
            }
        #pragma unroll 1
        for (int t = 0; t < 16; ++t) {
            const float bb = b1[t * 16 + l15];
            V8F C0, C1;
            #pragma unroll
            for (int r = 0; r < 8; ++r) { C0.f[r] = bb; C1.f[r] = bb; }
            #pragma unroll
            for (int c = 0; c < 4; ++c) {
                V16H B;   // shared by both row tiles
                const _Float16* wn =
                    W1T + (t * 16 + l15) * F_HID + c * 32 + hl * 8;
                B.f4[0] = *(const v4f*)wn;
                B.f4[1] = *(const v4f*)(wn + 16);
                C0.v = __builtin_amdgcn_wmma_f32_16x16x32_f16(
                    false, A1[0][c].v, false, B.v, (short)0, C0.v, false, false);
                C1.v = __builtin_amdgcn_wmma_f32_16x16x32_f16(
                    false, A1[1][c].v, false, B.v, (short)0, C1.v, false, false);
            }
            #pragma unroll
            for (int r = 0; r < 8; ++r) {
                int m0 = hl ? r + 8 : r;
                S.wij[m0 * F_OUT + t * 16 + l15] =
                    (_Float16)(C0.f[r] * S.phi[m0]);
                int m1 = 16 + m0;
                S.wij[m1 * F_OUT + t * 16 + l15] =
                    (_Float16)(C1.f[r] * S.phi[m1]);
            }
        }

        // ---- alpha[p,d,h] = sum_a Q_i*K_j*W_ij / sqrt(FH) -----------------
        #pragma unroll 1
        for (int u = 0; u < 2; ++u) {
            const int p = u * 16 + (lane >> 1), d = lane & 1;
            const int ia = S.ii[p], ja = S.jj[p];
            const float* Qr = Qws + ia * 256 + d * 128;
            const float* Kr = Kws + ja * 256 + d * 128;
            const float msk = pmask[p0 + p];
            #pragma unroll 1
            for (int h = 0; h < 8; ++h) {
                float acc = 0.f;
                #pragma unroll
                for (int a4 = 0; a4 < 4; ++a4) {
                    v4f q = *(const v4f*)(Qr + h * 16 + a4 * 4);
                    v4f k = *(const v4f*)(Kr + h * 16 + a4 * 4);
                    v4h w = *(const v4h*)(S.wij + p * F_OUT + d * 128 +
                                          h * 16 + a4 * 4);
                    acc += q.x * k.x * (float)w[0] + q.y * k.y * (float)w[1] +
                           q.z * k.z * (float)w[2] + q.w * k.w * (float)w[3];
                }
                S.alpha[p * 16 + d * 8 + h] = acc * 0.25f * msk;
            }
        }

        // ---- z = alpha * sph * x_j ; segment-sum via run-length + atomics --
        {
            float acc[8][4];
            int curI = -1;
            #pragma unroll 1
            for (int p = 0; p < 32; ++p) {
                const int i = S.ii[p];                 // wave-uniform
                if (i != curI) {
                    if (curI >= 0) flush_acc(out, curI, lane, acc);
                    curI = i;
                    #pragma unroll
                    for (int mm = 0; mm < 8; ++mm)
                        #pragma unroll
                        for (int s = 0; s < 4; ++s) acc[mm][s] = 0.f;
                }
                const int ja = S.jj[p];
                const float msk = pmask[p0 + p];
                float xj[4];
                #pragma unroll
                for (int s = 0; s < 4; ++s)
                    xj[s] = x[ja * F_DIM + lane + 32 * s] * msk;
                v4f sA = *(const v4f*)(sph + (p0 + p) * M_TOT);
                v4f sB = *(const v4f*)(sph + (p0 + p) * M_TOT + 4);
                float sphv[8] = {sA.x, sA.y, sA.z, sA.w,
                                 sB.x, sB.y, sB.z, sB.w};
                #pragma unroll
                for (int s = 0; s < 4; ++s) {
                    const int h = (lane + 32 * s) >> 4;
                    const float g0 = S.alpha[p * 16 + h]     * xj[s]; // l=1
                    const float g1 = S.alpha[p * 16 + 8 + h] * xj[s]; // l=2
                    #pragma unroll
                    for (int mm = 0; mm < 8; ++mm)             // repeats 3,5
                        acc[mm][s] += ((mm < 3) ? g0 : g1) * sphv[mm];
                }
            }
            if (curI >= 0) flush_acc(out, curI, lane, acc);
        }
    }
}

// ------------------------------------------------------------------- launch
extern "C" void kernel_launch(void* const* d_in, const int* in_sizes, int n_in,
                              void* d_out, int out_size, void* d_ws,
                              size_t ws_size, hipStream_t stream) {
    const float* x     = (const float*)d_in[0];
    const float* rbf   = (const float*)d_in[1];
    const float* sph   = (const float*)d_in[2];
    const float* phi   = (const float*)d_in[3];
    const int*   idx_i = (const int*)d_in[4];
    const int*   idx_j = (const int*)d_in[5];
    const float* pmask = (const float*)d_in[6];
    const float* Wq    = (const float*)d_in[7];
    const float* Wk    = (const float*)d_in[8];
    const float* W0    = (const float*)d_in[9];
    const float* b0    = (const float*)d_in[10];
    const float* W1    = (const float*)d_in[11];
    const float* b1    = (const float*)d_in[12];
    float* out = (float*)d_out;

    char* wsb = (char*)d_ws;
    float*    Qws = (float*)wsb;                          // 20,480,000 B
    float*    Kws = (float*)(wsb + 20480000);             // 20,480,000 B
    _Float16* W0T = (_Float16*)(wsb + 40960000);          //      8,192 B
    _Float16* W1T = (_Float16*)(wsb + 40960000 + 8192);   //     65,536 B

    k_zero<<<512, 256, 0, stream>>>(out, N_ATOMS * 1024);
    k_wt<<<64, 256, 0, stream>>>(W0, W1, W0T, W1T);
    k_qk<<<(N_ATOMS / 16 + 3) / 4, 128, 0, stream>>>(x, Wq, Wk, Qws, Kws);
    k_main<<<1024, 64, 0, stream>>>(rbf, sph, phi, idx_i, idx_j, pmask, b0,
                                    b1, x, Qws, Kws, W0T, W1T, out);
}
